// ComplexGraphAttentionLayer_61306363183804
// MI455X (gfx1250) — matrix operand order repaired and checked by reading, not compiled, not run on hardware
//
#include <hip/hip_runtime.h>

// =====================================================================
// Complex graph-attention layer for MI455X (gfx1250), wave32 + WMMA.
//
// Roofline: ~155 GFLOP of GEMM math vs ~0.2 GB of mandatory HBM traffic
// (~10 us at 23.3 TB/s) => memory-bound.  Strategy:
//   * convert every GEMM operand to bf16 ONCE (halves streamed bytes),
//   * all GEMMs via v_wmma_f32_16x16x32_bf16 (f32 accumulate),
//   * every fragment (A, B, P-from-LDS, V) loads as two b128 per lane:
//     V is written TRANSPOSED ([2D][tok]) by the v-projection epilogue
//     so PV B-fragments are contiguous like all others,
//   * flash-style fused attention: no SxS score tensor ever hits HBM,
//   * bias/CReLU/softmax/LN fused into producer epilogues.
// =====================================================================

#define DEV __device__ __forceinline__

typedef __attribute__((__ext_vector_type__(16))) __bf16    bf16x16;
typedef __attribute__((__ext_vector_type__(8)))  float     v8f;
typedef __attribute__((__ext_vector_type__(4)))  unsigned  u32x4;
typedef unsigned short ushort;

static constexpr int cB = 2, cS = 2048, cD = 512, cH = 8, cHD = 64;
static constexpr int cNT = cB * cS;            // total tokens
static constexpr float cEPS = 1e-5f;

// fragment <-> raw-dword view
union Frag {
  bf16x16  f;
  u32x4    q[2];
  unsigned u[8];
};

// f32 -> bf16 bits, round-to-nearest-even
DEV ushort f2bf_u(float x) {
  union { float f; unsigned u; } a; a.f = x;
  unsigned r = a.u + 0x7FFFu + ((a.u >> 16) & 1u);
  return (ushort)(r >> 16);
}
DEV unsigned pack2bf(float lo, float hi) {
  return (unsigned)f2bf_u(lo) | ((unsigned)f2bf_u(hi) << 16);
}

DEV v8f wmma_bf16(bf16x16 a, bf16x16 b, v8f c) {
  // (neg_a, A, neg_b, B, c_mod, C, reuse_a, reuse_b)
  return __builtin_amdgcn_wmma_f32_16x16x32_bf16(false, a, false, b,
                                                 (short)0, c, false, false);
}

// ---- bf16 fragment loaders (CDNA5 ISA 7.12.2 layouts) ----------------
// A (16x32): lane%16 = M; VGPR v pair k0 = (v/4)*16 + (lane>=16?8:0) + (v%4)*2
// => per lane two contiguous 16B runs: [khalf..khalf+7], [16+khalf..+7]
DEV bf16x16 ld_a_bf(const ushort* rowp, int lane) {
  const int khalf = (lane >= 16) ? 8 : 0;
  Frag r;
  r.q[0] = *(const u32x4*)(rowp + khalf);
  r.q[1] = *(const u32x4*)(rowp + 16 + khalf);
  return r.f;
}

// B (32x16): lane%16 = N; same K order as A fragment
// => per lane two contiguous 16B runs: [kh..kh+7], [16+kh..+7]
DEV bf16x16 ld_b_bf(const ushort* rowp, int lane) {
  const int kh = (lane >= 16) ? 8 : 0;
  Frag r;
  r.q[0] = *(const u32x4*)(rowp + kh);
  r.q[1] = *(const u32x4*)(rowp + 16 + kh);
  return r.f;
}

// =====================================================================
// One-shot f32 -> packed bf16 conversion (pair per thread, b32 stores)
// =====================================================================
__global__ __launch_bounds__(256) void cvt_kernel(
    const float* __restrict__ src, unsigned* __restrict__ dst, int npairs) {
  const int i = blockIdx.x * 256 + threadIdx.x;
  if (i < npairs) dst[i] = pack2bf(src[2 * i], src[2 * i + 1]);
}

// =====================================================================
// Complex linear on bf16 operands:
//   yr = xr*Wr^T - xi*Wi^T + br ;  yi = xr*Wi^T + xi*Wr^T + bi
// One wave = one 16x16 tile.  Output f32 (Yf) or bf16 (Yb), optional
// ReLU, optional transposed bf16 store ([2*Dout][N], used for V so the
// attention PV B-fragments become contiguous b128 loads).
// bf16 WMMA NEG bits are C-only (ISA 7.12) -> subtraction via 2nd accum.
// =====================================================================
__global__ __launch_bounds__(256) void cl_kernel(
    const ushort* __restrict__ X, const ushort* __restrict__ Wr,
    const ushort* __restrict__ Wi, const float* __restrict__ br,
    const float* __restrict__ bi, float* __restrict__ Yf,
    ushort* __restrict__ Yb, int N, int Din, int Dout, int relu,
    int trans) {
  const int lane = threadIdx.x & 31;
  const int wv   = threadIdx.x >> 5;
  const int tilesN = Dout >> 4;
  const int tilesM = N >> 4;
  const int tile = blockIdx.x * 8 + wv;             // wave-uniform guard
  if (tile >= tilesM * tilesN) return;
  const int tm = tile / tilesN, tn = tile % tilesN;
  const int r16 = lane & 15;
  const int half8 = (lane >= 16) ? 8 : 0;

  const ushort* xrow  = X  + (size_t)(tm * 16 + r16) * (size_t)(2 * Din);
  const ushort* wrrow = Wr + (size_t)(tn * 16 + r16) * (size_t)Din;
  const ushort* wirow = Wi + (size_t)(tn * 16 + r16) * (size_t)Din;

  v8f accP = {}, accM = {}, accQ = {};
  for (int kb = 0; kb < Din; kb += 32) {
    bf16x16 ar = ld_a_bf(xrow + kb, lane);
    bf16x16 ai = ld_a_bf(xrow + Din + kb, lane);
    bf16x16 wr = ld_b_bf(wrrow + kb, lane);
    bf16x16 wi = ld_b_bf(wirow + kb, lane);
    accP = wmma_bf16(ar, wr, accP);   // xr*Wr^T
    accM = wmma_bf16(ai, wi, accM);   // xi*Wi^T (subtracted later)
    accQ = wmma_bf16(ar, wi, accQ);   // xr*Wi^T
    accQ = wmma_bf16(ai, wr, accQ);   //  + xi*Wr^T
    __builtin_prefetch(wrrow + kb + 64, 0, 0);      // global_prefetch_b8
    __builtin_prefetch(wirow + kb + 64, 0, 0);
  }

  const int ncol = tn * 16 + r16;
  const float brv = br[ncol], biv = bi[ncol];
#pragma unroll
  for (int v = 0; v < 8; ++v) {
    const int row = tm * 16 + v + half8;            // C/D layout: M=v(+8)
    float yr = accP[v] - accM[v] + brv;
    float yi = accQ[v] + biv;
    if (relu) { yr = fmaxf(yr, 0.f); yi = fmaxf(yi, 0.f); }
    if (trans) {
      // Y^T: [2*Dout][N]; for fixed lane the 8 stores are consecutive rows
      Yb[(size_t)ncol * N + row]          = f2bf_u(yr);
      Yb[(size_t)(Dout + ncol) * N + row] = f2bf_u(yi);
    } else if (Yb) {
      const size_t rb = (size_t)row * (size_t)(2 * Dout);
      Yb[rb + ncol]        = f2bf_u(yr);
      Yb[rb + Dout + ncol] = f2bf_u(yi);
    } else {
      const size_t rb = (size_t)row * (size_t)(2 * Dout);
      Yf[rb + ncol]        = yr;
      Yf[rb + Dout + ncol] = yi;
    }
  }
}

// =====================================================================
// Fused complex attention, flash-style online softmax.
// Q/K are bf16 [B,S,2D]; V is bf16 transposed [2D][B*S].  One wave =
// 16-row q-tile of one (b,h); kv loop in 32-column tiles:
//   scores = (Re^2+Im^2)/sqrt(HD) + edge_bias[graph_mask]
//   online softmax (16-lane shfl_xor row reductions)
//   P staged through LDS (bf16) to re-read in A-fragment layout
//   PV accumulated with 8 WMMAs (contiguous b128 V-fragments thanks to
//   the transposed layout).  Output written bf16 (feeds o-proj GEMM).
// =====================================================================
__global__ __launch_bounds__(256) void attn_kernel(
    const ushort* __restrict__ Q, const ushort* __restrict__ K,
    const ushort* __restrict__ Vt, const int* __restrict__ GM,
    const float* __restrict__ EB, ushort* __restrict__ O) {
  __shared__ __align__(16) unsigned plsu[8 * 256];  // per-wave 16x32 bf16 P

  const int lane = threadIdx.x & 31;
  const int wv   = threadIdx.x >> 5;
  const int QT   = cS / 16;
  const int gidx = blockIdx.x * 8 + wv;
  if (gidx >= cB * cH * QT) return;                 // wave-uniform
  const int qt = gidx % QT;
  const int h  = (gidx / QT) % cH;
  const int b  = gidx / (QT * cH);
  const int r16 = lane & 15;
  const int half8 = (lane >= 16) ? 8 : 0;
  ushort* pw = (ushort*)plsu + wv * 512;

  float ebv[4];
#pragma unroll
  for (int e = 0; e < 4; ++e) ebv[e] = EB[e * cH + h];

  const size_t rowstride = (size_t)(2 * cD);
  const ushort* qrow = Q + ((size_t)b * cS + (size_t)(qt * 16 + r16)) * rowstride
                         + (size_t)h * cHD;
  bf16x16 qr0 = ld_a_bf(qrow,           lane);
  bf16x16 qr1 = ld_a_bf(qrow + 32,      lane);
  bf16x16 qi0 = ld_a_bf(qrow + cD,      lane);
  bf16x16 qi1 = ld_a_bf(qrow + cD + 32, lane);

  v8f Or[4] = {}, Oi[4] = {};
  float mrun[8], lrun[8];
#pragma unroll
  for (int v = 0; v < 8; ++v) { mrun[v] = -3.0e38f; lrun[v] = 0.f; }

  const ushort* kbaseh = K + (size_t)b * cS * rowstride + (size_t)h * cHD;
  // V^T rows: real plane row (h*HD+hd) at [.]*cNT, imag plane +cD rows
  const ushort* vtbase = Vt + (size_t)b * cS;       // + row*cNT + kv
  const int*    gmb    = GM + (size_t)b * cS * cS;

  for (int kv0 = 0; kv0 < cS; kv0 += 32) {
    float s[2][8];
#pragma unroll
    for (int c = 0; c < 2; ++c) {
      const ushort* krow = kbaseh + (size_t)(kv0 + c * 16 + r16) * rowstride;
      bf16x16 kr0 = ld_b_bf(krow,           lane);
      bf16x16 kr1 = ld_b_bf(krow + 32,      lane);
      bf16x16 ki0 = ld_b_bf(krow + cD,      lane);
      bf16x16 ki1 = ld_b_bf(krow + cD + 32, lane);
      v8f aR = {};                        // qr.kr + qi.ki
      aR = wmma_bf16(qr0, kr0, aR); aR = wmma_bf16(qr1, kr1, aR);
      aR = wmma_bf16(qi0, ki0, aR); aR = wmma_bf16(qi1, ki1, aR);
      v8f aI = {};                        // qr.ki
      aI = wmma_bf16(qr0, ki0, aI); aI = wmma_bf16(qr1, ki1, aI);
      v8f aJ = {};                        // qi.kr (subtracted)
      aJ = wmma_bf16(qi0, kr0, aJ); aJ = wmma_bf16(qi1, kr1, aJ);
      const int kcol = kv0 + c * 16 + r16;
#pragma unroll
      for (int v = 0; v < 8; ++v) {
        const int qri = qt * 16 + v + half8;
        const int e   = gmb[(size_t)qri * cS + kcol] & 3;
        const float im = aI[v] - aJ[v];
        s[c][v] = (aR[v] * aR[v] + im * im) * 0.125f + ebv[e];
      }
    }

    // --- online softmax over 32 new columns (row = 16-lane group) ---
    float alpha[8];
#pragma unroll
    for (int v = 0; v < 8; ++v) {
      float t = fmaxf(s[0][v], s[1][v]);
#pragma unroll
      for (int m = 1; m < 16; m <<= 1) t = fmaxf(t, __shfl_xor(t, m, 32));
      const float nm = fmaxf(mrun[v], t);
      alpha[v] = expf(mrun[v] - nm);
      const float p0 = expf(s[0][v] - nm);
      const float p1 = expf(s[1][v] - nm);
      float rs = p0 + p1;
#pragma unroll
      for (int m = 1; m < 16; m <<= 1) rs += __shfl_xor(rs, m, 32);
      lrun[v] = lrun[v] * alpha[v] + rs;
      mrun[v] = nm;
      pw[(v + half8) * 32 + r16]      = f2bf_u(p0);   // bf16 into LDS
      pw[(v + half8) * 32 + 16 + r16] = f2bf_u(p1);
    }
#pragma unroll
    for (int c = 0; c < 4; ++c)
#pragma unroll
      for (int v = 0; v < 8; ++v) { Or[c][v] *= alpha[v]; Oi[c][v] *= alpha[v]; }

    // P (16x32) re-read in A-fragment layout: two ds_load_b128 per lane
    bf16x16 pf = ld_a_bf(pw + r16 * 32, lane);

    // PV: V^T rows are contiguous along kv => two b128 loads per fragment
#pragma unroll
    for (int ch = 0; ch < 4; ++ch) {
      const ushort* vrrow =
          vtbase + (size_t)(h * cHD + ch * 16 + r16) * (size_t)cNT + kv0;
      const ushort* virow = vrrow + (size_t)cD * (size_t)cNT;
      bf16x16 vr = ld_b_bf(vrrow, lane);
      bf16x16 vi = ld_b_bf(virow, lane);
      Or[ch] = wmma_bf16(pf, vr, Or[ch]);
      Oi[ch] = wmma_bf16(pf, vi, Oi[ch]);
      __builtin_prefetch(vrrow + 32, 0, 0);
    }
    __builtin_prefetch(kbaseh + (size_t)(kv0 + 32 + r16) * rowstride, 0, 0);
  }

  // epilogue: normalize rows, write bf16 (consumed by o-projection GEMM)
#pragma unroll
  for (int v = 0; v < 8; ++v) {
    const float inv = 1.f / lrun[v];
    const int row = qt * 16 + v + half8;
    ushort* orow = O + ((size_t)b * cS + row) * rowstride + (size_t)h * cHD;
#pragma unroll
    for (int ch = 0; ch < 4; ++ch) {
      orow[ch * 16 + r16]      = f2bf_u(Or[ch][v] * inv);
      orow[cD + ch * 16 + r16] = f2bf_u(Oi[ch][v] * inv);
    }
  }
}

// =====================================================================
// Fused residual-add + LayerNorm; optional extra bf16 copy of the output
// (for when the next consumer is a GEMM).
// =====================================================================
__global__ __launch_bounds__(256) void ln_kernel(
    const float* __restrict__ R, const float* __restrict__ Yv,
    const float* __restrict__ w, const float* __restrict__ bp,
    float* __restrict__ out, ushort* __restrict__ outb, int width) {
  __shared__ float s1[256], s2[256];
  const int row = blockIdx.x, t = threadIdx.x;
  const float* r0 = R  + (size_t)row * width;
  const float* y0 = Yv + (size_t)row * width;
  float a = 0.f, b2 = 0.f;
  for (int j = t; j < width; j += 256) {
    const float v = r0[j] + y0[j];
    a += v; b2 += v * v;
  }
  s1[t] = a; s2[t] = b2; __syncthreads();
  for (int off = 128; off > 0; off >>= 1) {
    if (t < off) { s1[t] += s1[t + off]; s2[t] += s2[t + off]; }
    __syncthreads();
  }
  const float mean = s1[0] / width;
  const float var  = s2[0] / width - mean * mean;
  const float inv  = rsqrtf(var + cEPS);
  for (int j = t; j < width; j += 256) {
    const float v = (r0[j] + y0[j] - mean) * inv * w[j] + bp[j];
    out[(size_t)row * width + j] = v;
    if (outb) outb[(size_t)row * width + j] = f2bf_u(v);
  }
}

// =====================================================================
// Host launcher.  Input flattening (setup_inputs dict order):
//  0:x  1:graph_mask  2..5:q{Wr,Wi,br,bi}  6..9:k  10..13:v  14..17:o
//  18..21:ffn1  22..25:ffn2  26:edge_bias  27:norm1_w 28:norm1_b
//  29:norm2_w 30:norm2_b
// =====================================================================
extern "C" void kernel_launch(void* const* d_in, const int* in_sizes, int n_in,
                              void* d_out, int out_size, void* d_ws,
                              size_t ws_size, hipStream_t stream) {
  (void)in_sizes; (void)n_in; (void)out_size; (void)ws_size;
  const float* x  = (const float*)d_in[0];
  const int*   gm = (const int*)d_in[1];
  const float* eb = (const float*)d_in[26];

  const size_t SZ  = (size_t)cB * cS * 2 * cD;   // activation elements
  const size_t WSQ = (size_t)cD * cD;            // square weight elements
  const size_t WFF = (size_t)(4 * cD) * cD;      // ffn weight elements

  // bump allocator over workspace (256B aligned chunks)
  char* base = (char*)d_ws;
  size_t off = 0;
  auto alloc = [&](size_t bytes) -> void* {
    void* p = base + off;
    off += (bytes + 255) & ~(size_t)255;
    return p;
  };
  ushort* xb    = (ushort*)alloc(SZ * 2);        // bf16 x
  ushort* wbf[12];                               // bf16 weights, d_in idx:
  const int widx[12]  = {2, 3, 6, 7, 10, 11, 14, 15, 18, 19, 22, 23};
  const size_t wsz[12] = {WSQ, WSQ, WSQ, WSQ, WSQ, WSQ, WSQ, WSQ,
                          WFF, WFF, WFF, WFF};
  for (int i = 0; i < 12; ++i) wbf[i] = (ushort*)alloc(wsz[i] * 2);
  ushort* qb    = (ushort*)alloc(SZ * 2);
  ushort* kb    = (ushort*)alloc(SZ * 2);
  ushort* vtb   = (ushort*)alloc(SZ * 2);        // V transposed [2D][B*S]
  ushort* attnb = (ushort*)alloc(SZ * 2);
  float*  oproj = (float*)alloc(SZ * 4);
  float*  x1    = (float*)alloc(SZ * 4);
  ushort* x1b   = (ushort*)alloc(SZ * 2);
  ushort* h1    = qb;          // 4*SZ bf16, reuses q/k/v/attn (dead)
  float*  h2    = oproj;       // reuses oproj after LN1

  // ---- one-shot bf16 conversions -------------------------------------
  auto cvt = [&](const float* s, ushort* d, size_t n) {
    const int np = (int)(n / 2);
    cvt_kernel<<<dim3((np + 255) / 256), dim3(256), 0, stream>>>(
        s, (unsigned*)d, np);
  };
  cvt(x, xb, SZ);
  for (int i = 0; i < 12; ++i) cvt((const float*)d_in[widx[i]], wbf[i], wsz[i]);

  const int N = cNT;
  auto cl = [&](const ushort* X, int wi, int bi0, float* Yf, ushort* Yb,
                int Din, int Dout, int relu, int trans) {
    const int tiles = (N / 16) * (Dout / 16);
    cl_kernel<<<dim3((tiles + 7) / 8), dim3(256), 0, stream>>>(
        X, wbf[wi], wbf[wi + 1], (const float*)d_in[bi0],
        (const float*)d_in[bi0 + 1], Yf, Yb, N, Din, Dout, relu, trans);
  };

  cl(xb, 0, 4,  nullptr, qb,  cD, cD, 0, 0);        // q projection -> bf16
  cl(xb, 2, 8,  nullptr, kb,  cD, cD, 0, 0);        // k projection -> bf16
  cl(xb, 4, 12, nullptr, vtb, cD, cD, 0, 1);        // v projection -> bf16^T
  attn_kernel<<<dim3((cB * cH * (cS / 16)) / 8), dim3(256), 0, stream>>>(
      qb, kb, vtb, gm, eb, attnb);                  // fused attention -> bf16
  cl(attnb, 6, 16, oproj, nullptr, cD, cD, 0, 0);   // output projection -> f32
  ln_kernel<<<dim3(N), dim3(256), 0, stream>>>(     // residual + LN1
      x, oproj, (const float*)d_in[27], (const float*)d_in[28],
      x1, x1b, 2 * cD);
  cl(x1b, 8, 20, nullptr, h1, cD, 4 * cD, 1, 0);    // ffn1 + CReLU -> bf16
  cl(h1, 10, 24, h2, nullptr, 4 * cD, cD, 0, 0);    // ffn2 -> f32
  ln_kernel<<<dim3(N), dim3(256), 0, stream>>>(     // residual + LN2
      x1, h2, (const float*)d_in[29], (const float*)d_in[30],
      (float*)d_out, nullptr, 2 * cD);
}
